// UMixer_78417512890552
// MI455X (gfx1250) — matrix operand order repaired and checked by reading, not compiled, not run on hardware
//
#include <hip/hip_runtime.h>
#include <math.h>

// ============================================================================
// UMixer forward for MI455X (gfx1250, wave32).
// All GEMMs via v_wmma_f32_16x16x32_bf16 with bf16 hi/lo split (fp32-quality),
// 2x2 register tiling (12 WMMAs per K-step per wave).
// Operands pre-packed into the ISA wave32 WMMA fragment layouts.
// ============================================================================

typedef __bf16 bf16x16 __attribute__((ext_vector_type(16)));
typedef float  f32x8   __attribute__((ext_vector_type(8)));

#define DEVFN static __device__ __forceinline__

DEVFN unsigned short f2bf(float f) {           // round-to-nearest-even fp32->bf16
  unsigned u = __float_as_uint(f);
  unsigned r = ((u >> 16) & 1u) + 0x7fffu;
  return (unsigned short)((u + r) >> 16);
}
DEVFN float bf2f(unsigned short h) { return __uint_as_float(((unsigned)h) << 16); }
DEVFN float gelu_exact(float x) { return 0.5f * x * (1.0f + erff(x * 0.70710678118654752440f)); }

// ---------------------------------------------------------------------------
// Pack A (M x K, fp32, generic strides) -> WMMA A-fragments (16x32 bf16 tiles)
// Fragment = 32 lanes x 16 bf16, lane-contiguous.  hi/lo split arrays.
// A layout (ISA 7.12.2): lane: m = lane&15, half = lane>>4;
//   slot j16 (0..15): v = j16>>1; k = (v>>2)*16 + half*8 + (v&3)*2 + (j16&1)
// ---------------------------------------------------------------------------
__global__ void __launch_bounds__(256)
k_packA(const float* __restrict__ src, unsigned short* __restrict__ hi,
        unsigned short* __restrict__ lo, int batches,
        long long bS, long long rS, long long cS, int M, int K, int tM, int kT) {
  long long tid = (long long)blockIdx.x * 256 + threadIdx.x;
  long long total = (long long)batches * tM * kT * 64;
  if (tid >= total) return;
  int lane = (int)(tid & 31);
  int plane = (int)((tid >> 5) & 1);
  long long frag = tid >> 6;
  int kt = (int)(frag % kT);
  int tm = (int)((frag / kT) % tM);
  int b  = (int)(frag / ((long long)kT * tM));
  int m = lane & 15, half = lane >> 4;
  int mg = tm * 16 + m;
  long long outBase = frag * 512 + (long long)lane * 16 + plane * 8;
#pragma unroll
  for (int j = 0; j < 8; ++j) {
    int j16 = plane * 8 + j;
    int v = j16 >> 1;
    int k = (v >> 2) * 16 + half * 8 + (v & 3) * 2 + (j16 & 1);
    int kg = kt * 32 + k;
    float val = 0.0f;
    if (mg < M && kg < K)
      val = src[(long long)b * bS + (long long)mg * rS + (long long)kg * cS];
    unsigned short h = f2bf(val);
    hi[outBase + j] = h;
    lo[outBase + j] = f2bf(val - bf2f(h));
  }
}

// ---------------------------------------------------------------------------
// Pack B (K x N, fp32, generic strides) -> WMMA B-fragments (32x16 bf16 tiles)
// B layout: lane: n = lane&15, half = lane>>4; slot j16: k = (j16>>3)*16 + half*8 + (j16&7)
// ---------------------------------------------------------------------------
__global__ void __launch_bounds__(256)
k_packB(const float* __restrict__ src, unsigned short* __restrict__ hi,
        unsigned short* __restrict__ lo, int batches,
        long long bS, long long rS, long long cS, int K, int N, int kT, int tN) {
  long long tid = (long long)blockIdx.x * 256 + threadIdx.x;
  long long total = (long long)batches * kT * tN * 64;
  if (tid >= total) return;
  int lane = (int)(tid & 31);
  int plane = (int)((tid >> 5) & 1);
  long long frag = tid >> 6;
  int tn = (int)(frag % tN);
  int kt = (int)((frag / tN) % kT);
  int b  = (int)(frag / ((long long)tN * kT));
  int n = lane & 15, half = lane >> 4;
  int ng = tn * 16 + n;
  long long outBase = frag * 512 + (long long)lane * 16 + plane * 8;
#pragma unroll
  for (int j = 0; j < 8; ++j) {
    int j16 = plane * 8 + j;
    int kg = kt * 32 + (j16 >> 3) * 16 + half * 8 + (j16 & 7);
    float val = 0.0f;
    if (kg < K && ng < N)
      val = src[(long long)b * bS + (long long)kg * rS + (long long)ng * cS];
    unsigned short h = f2bf(val);
    hi[outBase + j] = h;
    lo[outBase + j] = f2bf(val - bf2f(h));
  }
}

// ---------------------------------------------------------------------------
// Batched WMMA GEMM with split-K and WMxWN register tiling (one wave per
// (16*WM)x(16*WN) C super-tile).  Per 16x16 sub-tile:
//   acc += Ahi*Bhi + Ahi*Blo + Alo*Bhi   (fp32-quality bf16 emulation)
// Requires tM % WM == 0 and tN % WN == 0 (host pads).
// ---------------------------------------------------------------------------
template <int WM, int WN>
__global__ void __launch_bounds__(256)
k_wmma_gemm(const unsigned short* __restrict__ Ahi, const unsigned short* __restrict__ Alo,
            const unsigned short* __restrict__ Bhi, const unsigned short* __restrict__ Blo,
            float* __restrict__ C, int batches, int tM, int tN, int kT,
            int splitK, int kPer) {
  int wave = blockIdx.x * 8 + (threadIdx.x >> 5);
  int lane = threadIdx.x & 31;
  int tMh = tM / WM, tNh = tN / WN;
  long long totalWaves = (long long)batches * splitK * tMh * tNh;
  if (wave >= totalWaves) return;
  int tn0 = (wave % tNh) * WN;
  int tm0 = ((wave / tNh) % tMh) * WM;
  int bs = wave / (tNh * tMh);
  int s  = bs % splitK;
  int b  = bs / splitK;
  int k0 = s * kPer;
  int k1 = k0 + kPer; if (k1 > kT) k1 = kT;

  const bf16x16* Afh = (const bf16x16*)Ahi;
  const bf16x16* Afl = (const bf16x16*)Alo;
  const bf16x16* Bfh = (const bf16x16*)Bhi;
  const bf16x16* Bfl = (const bf16x16*)Blo;

  long long aBase = (long long)b * tM;          // in 16-row tiles
  long long bBase = (long long)b * kT;

  f32x8 acc[WM][WN];
#pragma unroll
  for (int i = 0; i < WM; ++i)
#pragma unroll
    for (int j = 0; j < WN; ++j)
      acc[i][j] = (f32x8){0.f, 0.f, 0.f, 0.f, 0.f, 0.f, 0.f, 0.f};

  for (int kt = k0; kt < k1; ++kt) {
    long long ai[WM], bi[WN];
#pragma unroll
    for (int i = 0; i < WM; ++i)
      ai[i] = ((aBase + tm0 + i) * kT + kt) * 32 + lane;
#pragma unroll
    for (int j = 0; j < WN; ++j)
      bi[j] = ((bBase + kt) * tN + tn0 + j) * 32 + lane;
    if (kt + 8 < k1) {  // stream next tiles into cache (global_prefetch_b8)
      __builtin_prefetch((const void*)(Afh + ai[0] + 8 * 32), 0, 1);
      __builtin_prefetch((const void*)(Bfh + bi[0] + 8 * (long long)tN * 32), 0, 1);
    }
    bf16x16 ah[WM], al[WM], bh[WN], bl[WN];
#pragma unroll
    for (int i = 0; i < WM; ++i) { ah[i] = Afh[ai[i]]; al[i] = Afl[ai[i]]; }
#pragma unroll
    for (int j = 0; j < WN; ++j) { bh[j] = Bfh[bi[j]]; bl[j] = Bfl[bi[j]]; }
#pragma unroll
    for (int i = 0; i < WM; ++i) {
#pragma unroll
      for (int j = 0; j < WN; ++j) {
        acc[i][j] = __builtin_amdgcn_wmma_f32_16x16x32_bf16(false, ah[i], false, bh[j],
                                                            (short)0, acc[i][j], false, false);
        acc[i][j] = __builtin_amdgcn_wmma_f32_16x16x32_bf16(false, ah[i], false, bl[j],
                                                            (short)0, acc[i][j], false, false);
        acc[i][j] = __builtin_amdgcn_wmma_f32_16x16x32_bf16(false, al[i], false, bh[j],
                                                            (short)0, acc[i][j], false, false);
      }
    }
  }
  // C layout (ISA): VGPR r, lane L: m = r + 8*(L>>4), n = L&15
  int Mp = tM * 16, Np = tN * 16;
  long long outBase = ((long long)(b * splitK + s)) * Mp * Np;
#pragma unroll
  for (int i = 0; i < WM; ++i) {
#pragma unroll
    for (int j = 0; j < WN; ++j) {
      int col = (tn0 + j) * 16 + (lane & 15);
      int rowBase = (tm0 + i) * 16 + (lane >> 4) * 8;
#pragma unroll
      for (int r = 0; r < 8; ++r)
        C[outBase + (long long)(rowBase + r) * Np + col] = acc[i][j][r];
    }
  }
}

// ---------------------------------------------------------------------------
// RevIN stats + normalized, transposed series Xn[bn, l]
// ---------------------------------------------------------------------------
__global__ void __launch_bounds__(256)
k_norm_stats(const float* __restrict__ hist, const float* __restrict__ rw,
             const float* __restrict__ rb, float* __restrict__ meanp,
             float* __restrict__ stdp, float* __restrict__ Xn) {
  __shared__ float s1[256], s2[256];
  int bn = blockIdx.x, b = bn >> 6, n = bn & 63;
  float a = 0.f, q = 0.f;
  for (int l = threadIdx.x; l < 720; l += 256) {
    float v = hist[((long long)(b * 720 + l)) * 64 + n];
    a += v; q += v * v;
  }
  s1[threadIdx.x] = a; s2[threadIdx.x] = q; __syncthreads();
  for (int off = 128; off > 0; off >>= 1) {
    if (threadIdx.x < (unsigned)off) { s1[threadIdx.x] += s1[threadIdx.x + off]; s2[threadIdx.x] += s2[threadIdx.x + off]; }
    __syncthreads();
  }
  float mean = s1[0] * (1.0f / 720.f);
  float var  = s2[0] * (1.0f / 720.f) - mean * mean;
  float sd   = sqrtf(var + 1e-5f);
  if (threadIdx.x == 0) { meanp[bn] = mean; stdp[bn] = sd; }
  float w = rw[n], bb = rb[n];
  for (int l = threadIdx.x; l < 720; l += 256) {
    float v = hist[((long long)(b * 720 + l)) * 64 + n];
    Xn[(long long)bn * 720 + l] = (v - mean) / sd * w + bb;
  }
}

__global__ void k_posemb(float* __restrict__ pe) {
  int idx = blockIdx.x * blockDim.x + threadIdx.x;
  if (idx >= 180 * 256) return;
  int p = idx >> 8, d = idx & 255, i = d >> 1;
  float div = expf((float)(2 * i) * (-logf(10000.f) / 256.f));
  float ang = (float)p * div;
  pe[idx] = (d & 1) ? cosf(ang) : sinf(ang);
}

// patch embed: mode0 -> from xraw(256x1440)+pred_b (pad-repeat last), P=180
//              mode1 -> from history transposed (pad-repeat last), P=90
__global__ void __launch_bounds__(256)
k_embed(const float* __restrict__ xraw, const float* __restrict__ predb,
        const float* __restrict__ hist, const float* __restrict__ embw,
        const float* __restrict__ pe, float* __restrict__ out, int P, int mode) {
  long long idx = (long long)blockIdx.x * 256 + threadIdx.x;
  long long total = 256LL * P * 256;
  if (idx >= total) return;
  int d = (int)(idx & 255);
  long long r = idx >> 8;
  int p = (int)(r % P);
  int bn = (int)(r / P);
  float s = 0.f;
#pragma unroll
  for (int k = 0; k < 16; ++k) {
    int c = p * 8 + k;
    float v;
    if (mode == 0) {
      int cc = c < 1440 ? c : 1439;
      v = xraw[(long long)bn * 1440 + cc] + predb[cc];
    } else {
      int l = c < 720 ? c : 719;
      int b = bn >> 6, n = bn & 63;
      v = hist[((long long)(b * 720 + l)) * 64 + n];
    }
    s += v * embw[k * 256 + d];
  }
  out[((long long)bn * P + p) * 256 + d] = s + pe[p * 256 + d];
}

__global__ void __launch_bounds__(256)
k_bias_gelu(float* __restrict__ C, const float* __restrict__ bias,
            int batches, int M, int Np, int N, int bbs) {
  long long idx = (long long)blockIdx.x * 256 + threadIdx.x;
  long long total = (long long)batches * M * Np;
  if (idx >= total) return;
  int n = (int)(idx % Np);
  long long r = idx / Np;
  int m = (int)(r % M);
  int b = (int)(r / M);
  if (n >= N) return;
  long long o = ((long long)b * M + m) * Np + n;
  C[o] = gelu_exact(C[o] + bias[(long long)b * bbs + n]);
}

// temp_mix LN: out[bn,p,e] = LN_e( O[p,bn,e] + b2[p,e] + X[bn,p,e] )
__global__ void __launch_bounds__(256)
k_temp_ln(const float* __restrict__ O, const float* __restrict__ b2,
          const float* __restrict__ X, const float* __restrict__ nw,
          const float* __restrict__ nb, float* __restrict__ out) {
  __shared__ float s1[256], s2[256];
  int row = blockIdx.x;              // 256*180 rows
  int bn = row / 180, p = row % 180;
  int e = threadIdx.x;
  float v = O[((long long)p * 256 + bn) * 256 + e] + b2[(long long)p * 256 + e] +
            X[((long long)bn * 180 + p) * 256 + e];
  s1[e] = v; s2[e] = v * v; __syncthreads();
  for (int off = 128; off > 0; off >>= 1) {
    if (e < off) { s1[e] += s1[e + off]; s2[e] += s2[e + off]; }
    __syncthreads();
  }
  float mu = s1[0] * (1.f / 256.f);
  float var = s2[0] * (1.f / 256.f) - mu * mu;
  float inv = rsqrtf(var + 1e-5f);
  out[((long long)bn * 180 + p) * 256 + e] = (v - mu) * inv * nw[e] + nb[e];
}

// chan_mix LN: out[bn,q,d] = LN_d( O[d,bn,q] + b2[d,q] + X[bn,q,d] ), Np=192
__global__ void __launch_bounds__(256)
k_chan_ln(const float* __restrict__ O, const float* __restrict__ b2,
          const float* __restrict__ X, const float* __restrict__ nw,
          const float* __restrict__ nb, float* __restrict__ out) {
  __shared__ float s1[256], s2[256];
  int row = blockIdx.x;
  int bn = row / 180, q = row % 180;
  int d = threadIdx.x;
  float v = O[(long long)d * (256 * 192) + (long long)bn * 192 + q] +
            b2[(long long)d * 180 + q] +
            X[((long long)bn * 180 + q) * 256 + d];
  s1[d] = v; s2[d] = v * v; __syncthreads();
  for (int off = 128; off > 0; off >>= 1) {
    if (d < off) { s1[d] += s1[d + off]; s2[d] += s2[d + off]; }
    __syncthreads();
  }
  float mu = s1[0] * (1.f / 256.f);
  float var = s2[0] * (1.f / 256.f) - mu * mu;
  float inv = rsqrtf(var + 1e-5f);
  out[((long long)bn * 180 + q) * 256 + d] = (v - mu) * inv * nw[d] + nb[d];
}

// plain LN over d for rows p < Ptake of a (bn, Psrc, 256) tensor
__global__ void __launch_bounds__(256)
k_ln_plain(const float* __restrict__ src, float* __restrict__ dst,
           int Psrc, int Ptake, const float* __restrict__ w, const float* __restrict__ b) {
  __shared__ float s1[256], s2[256];
  int row = blockIdx.x;
  int bn = row / Ptake, p = row % Ptake;
  int d = threadIdx.x;
  float v = src[((long long)bn * Psrc + p) * 256 + d];
  s1[d] = v; s2[d] = v * v; __syncthreads();
  for (int off = 128; off > 0; off >>= 1) {
    if (d < off) { s1[d] += s1[d + off]; s2[d] += s2[d + off]; }
    __syncthreads();
  }
  float mu = s1[0] * (1.f / 256.f);
  float var = s2[0] * (1.f / 256.f) - mu * mu;
  float inv = rsqrtf(var + 1e-5f);
  dst[((long long)bn * Ptake + p) * 256 + d] = (v - mu) * inv * w[d] + b[d];
}

__global__ void k_comb(float* __restrict__ X, const float* __restrict__ cw,
                       const float* __restrict__ cb, long long ntot) {
  long long i = (long long)blockIdx.x * 256 + threadIdx.x;
  if (i < ntot) X[i] = X[i] * cw[0] + cb[0];
}

// s-correction: circular autocorrelation (= irfft_ortho(F conj F) * sqrt(n))
__global__ void __launch_bounds__(128)
k_scorr(const float* __restrict__ a, const float* __restrict__ b, float* __restrict__ scale) {
  __shared__ float sa[90], sb[90], red[128], red2[128];
  int blk = blockIdx.x;
  int d = blk & 255, bn = blk >> 8;
  int t = threadIdx.x;
  if (t < 90) {
    sa[t] = a[((long long)(bn * 90 + t)) * 256 + d];
    sb[t] = b[((long long)(bn * 90 + t)) * 256 + d];
  }
  __syncthreads();
  float xa = 0.f, xb = 0.f;
  if (t < 90) {
    float s0 = 0.f, s1 = 0.f;
    for (int j = 0; j < 90; ++j) {
      int jj = j + t; if (jj >= 90) jj -= 90;
      s0 += sa[j] * sa[jj];
      s1 += sb[j] * sb[jj];
    }
    const float rs = 0.10540925533894598f;  // 1/sqrt(90) (ortho-norm factor)
    xa = fmaxf(s0 * rs, 0.f);
    xb = fmaxf(s1 * rs, 0.f);
  }
  red[t] = xa * xb; red2[t] = xb * xb; __syncthreads();
  for (int off = 64; off > 0; off >>= 1) {
    if (t < off) { red[t] += red[t + off]; red2[t] += red2[t + off]; }
    __syncthreads();
  }
  if (t == 0) scale[(long long)bn * 256 + d] = sqrtf(red[0] / (red2[0] + 0.001f));
}

// xct[bn, d*180+p] = scale[bn,d] * X[bn,p,d]
__global__ void __launch_bounds__(256)
k_scale_transpose(const float* __restrict__ X, const float* __restrict__ scale,
                  float* __restrict__ xct) {
  long long idx = (long long)blockIdx.x * 256 + threadIdx.x;
  if (idx >= 11796480LL) return;
  int d = (int)(idx & 255);
  long long r = idx >> 8;
  int p = (int)(r % 180);
  int bn = (int)(r / 180);
  xct[(long long)bn * 46080 + (long long)d * 180 + p] =
      X[idx] * scale[(long long)bn * 256 + d];
}

// sum split-K partials laid out as (splitK, M, Np); emit compact (M, N)
__global__ void __launch_bounds__(256)
k_reduce_splitk(const float* __restrict__ Cp, float* __restrict__ C,
                int splitK, int M, int Np, int N) {
  long long idx = (long long)blockIdx.x * 256 + threadIdx.x;
  long long total = (long long)M * N;
  if (idx >= total) return;
  int n = (int)(idx % N);
  int m = (int)(idx / N);
  float s = 0.f;
  for (int k = 0; k < splitK; ++k)
    s += Cp[((long long)k * M + m) * Np + n];
  C[idx] = s;
}

__global__ void __launch_bounds__(256)
k_head_final(const float* __restrict__ C, const float* __restrict__ hb,
             const float* __restrict__ rw, const float* __restrict__ rb,
             const float* __restrict__ meanp, const float* __restrict__ stdp,
             float* __restrict__ out) {
  long long idx = (long long)blockIdx.x * 256 + threadIdx.x;
  if (idx >= 184320LL) return;
  int t = (int)(idx % 720);
  int bn = (int)(idx / 720);
  int b = bn >> 6, n = bn & 63;
  float v = C[(long long)bn * 720 + t] + hb[t];
  v = (v - rb[n]) / (rw[n] + 1e-10f) * stdp[bn] + meanp[bn];
  out[((long long)(b * 720 + t)) * 64 + n] = v;
}

// ============================================================================
// Host launcher
// ============================================================================
extern "C" void kernel_launch(void* const* d_in, const int* in_sizes, int n_in,
                              void* d_out, int out_size, void* d_ws, size_t ws_size,
                              hipStream_t stream) {
  (void)in_sizes; (void)n_in; (void)out_size;
  const float* hist  = (const float*)d_in[0];
  const float* revw  = (const float*)d_in[5];
  const float* revb  = (const float*)d_in[6];
  const float* predw = (const float*)d_in[7];
  const float* predb = (const float*)d_in[8];
  const float* embw  = (const float*)d_in[9];
  const float* lnw   = (const float*)d_in[10];
  const float* lnb   = (const float*)d_in[11];
  const float* combw = (const float*)d_in[12];
  const float* combb = (const float*)d_in[13];
  const float* headw = (const float*)d_in[14];
  const float* headb = (const float*)d_in[15];
  const float* tmd_w1 = (const float*)d_in[16];
  const float* tmd_b1 = (const float*)d_in[17];
  const float* tmd_w2 = (const float*)d_in[18];
  const float* tmd_b2 = (const float*)d_in[19];
  const float* tmd_nw = (const float*)d_in[20];
  const float* tmd_nb = (const float*)d_in[21];
  const float* tmu_w1 = (const float*)d_in[22];
  const float* tmu_b1 = (const float*)d_in[23];
  const float* tmu_w2 = (const float*)d_in[24];
  const float* tmu_b2 = (const float*)d_in[25];
  const float* tmu_nw = (const float*)d_in[26];
  const float* tmu_nb = (const float*)d_in[27];
  const float* cmd_w1 = (const float*)d_in[28];
  const float* cmd_b1 = (const float*)d_in[29];
  const float* cmd_w2 = (const float*)d_in[30];
  const float* cmd_b2 = (const float*)d_in[31];
  const float* cmd_nw = (const float*)d_in[32];
  const float* cmd_nb = (const float*)d_in[33];
  const float* cmu_w1 = (const float*)d_in[34];
  const float* cmu_b1 = (const float*)d_in[35];
  const float* cmu_w2 = (const float*)d_in[36];
  const float* cmu_b2 = (const float*)d_in[37];
  const float* cmu_nw = (const float*)d_in[38];
  const float* cmu_nb = (const float*)d_in[39];

  // ----- workspace carve (bump allocator, 256B aligned) -----
  char* wsp = (char*)d_ws;
  auto alloc = [&](size_t bytes) -> char* {
    char* p = wsp; wsp += (bytes + 255) & ~(size_t)255; return p;
  };
  float* meanp  = (float*)alloc(256 * 4);
  float* stdp   = (float*)alloc(256 * 4);
  float* pe     = (float*)alloc(180 * 256 * 4);
  float* Xn     = (float*)alloc(256 * 720 * 4);
  float* xraw   = (float*)alloc(256 * 1440 * 4);
  float* scalep = (float*)alloc(65536 * 4);
  float* headC  = (float*)alloc(256 * 720 * 4);
  float* X      = (float*)alloc(11796480LL * 4);   // activations (bn,180,256)
  float* Y      = (float*)alloc(11796480LL * 4);   // ping-pong
  float* G1     = (float*)alloc(50331648LL);       // GEMM intermediate / xct
  float* G2     = (float*)alloc(50331648LL);       // GEMM intermediate / splitK partials
  unsigned short* PAhi = (unsigned short*)alloc(25165824LL);
  unsigned short* PAlo = (unsigned short*)alloc(25165824LL);
  unsigned short* PBhi = (unsigned short*)alloc(67829760LL);  // head: 1440*46 frags
  unsigned short* PBlo = (unsigned short*)alloc(67829760LL);
  float* xold   = (float*)alloc(5898240LL * 4);
  float* xoldln = (float*)alloc(5898240LL * 4);
  float* xcln   = (float*)alloc(5898240LL * 4);
  if ((size_t)(wsp - (char*)d_ws) > ws_size) return;  // insufficient scratch

  auto packA = [&](const float* src, int batches, long long bS, long long rS,
                   long long cS, int M, int K, int tM, int kT) {
    long long total = (long long)batches * tM * kT * 64;
    k_packA<<<(unsigned)((total + 255) / 256), 256, 0, stream>>>(
        src, PAhi, PAlo, batches, bS, rS, cS, M, K, tM, kT);
  };
  auto packB = [&](const float* src, int batches, long long bS, long long rS,
                   long long cS, int K, int N, int kT, int tN) {
    long long total = (long long)batches * kT * tN * 64;
    k_packB<<<(unsigned)((total + 255) / 256), 256, 0, stream>>>(
        src, PBhi, PBlo, batches, bS, rS, cS, K, N, kT, tN);
  };
  auto gemm = [&](float* C, int batches, int tM, int tN, int kT, int splitK, int kPer) {
    long long waves = (long long)batches * splitK * (tM / 2) * (tN / 2);
    k_wmma_gemm<2, 2><<<(unsigned)((waves + 7) / 8), 256, 0, stream>>>(
        PAhi, PAlo, PBhi, PBlo, C, batches, tM, tN, kT, splitK, kPer);
  };

  // ----- RevIN + pred matmul + patch embeds -----
  k_norm_stats<<<256, 256, 0, stream>>>(hist, revw, revb, meanp, stdp, Xn);
  k_posemb<<<(180 * 256 + 255) / 256, 256, 0, stream>>>(pe);
  packA(Xn, 1, 0, 720, 1, 256, 720, 16, 23);
  packB(predw, 1, 0, 1440, 1, 720, 1440, 23, 90);
  gemm(xraw, 1, 16, 90, 23, 1, 23);
  k_embed<<<(unsigned)((256LL * 180 * 256 + 255) / 256), 256, 0, stream>>>(
      xraw, predb, hist, embw, pe, X, 180, 0);
  k_embed<<<(unsigned)((256LL * 90 * 256 + 255) / 256), 256, 0, stream>>>(
      xraw, predb, hist, embw, pe, xold, 90, 1);

  float* Xc = X;   // current activation
  float* Xo = Y;   // output of next mix

  // temp_mix: 180 batched (256x256)@(256x256) GEMMs x2 + GELU + residual LN
  auto temp_mix = [&](const float* w1, const float* b1, const float* w2,
                      const float* b2, const float* nw, const float* nb) {
    packA(Xc, 180, 256, 46080, 1, 256, 256, 16, 8);
    packB(w1, 180, 65536, 256, 1, 256, 256, 8, 16);
    gemm(G1, 180, 16, 16, 8, 1, 8);
    k_bias_gelu<<<(unsigned)((180LL * 256 * 256 + 255) / 256), 256, 0, stream>>>(
        G1, b1, 180, 256, 256, 256, 256);
    packA(G1, 180, 65536, 256, 1, 256, 256, 16, 8);
    packB(w2, 180, 65536, 256, 1, 256, 256, 8, 16);
    gemm(G2, 180, 16, 16, 8, 1, 8);
    k_temp_ln<<<256 * 180, 256, 0, stream>>>(G2, b2, Xc, nw, nb, Xo);
    float* t = Xc; Xc = Xo; Xo = t;
  };
  // chan_mix: 256 batched (256x180)@(180x180) GEMMs (K,N padded to 192)
  auto chan_mix = [&](const float* w1, const float* b1, const float* w2,
                      const float* b2, const float* nw, const float* nb) {
    packA(Xc, 256, 1, 46080, 256, 256, 180, 16, 6);
    packB(w1, 256, 32400, 180, 1, 180, 180, 6, 12);
    gemm(G1, 256, 16, 12, 6, 1, 6);
    k_bias_gelu<<<(unsigned)((256LL * 256 * 192 + 255) / 256), 256, 0, stream>>>(
        G1, b1, 256, 256, 192, 180, 180);
    packA(G1, 256, 49152, 192, 1, 256, 180, 16, 6);
    packB(w2, 256, 32400, 180, 1, 180, 180, 6, 12);
    gemm(G2, 256, 16, 12, 6, 1, 6);
    k_chan_ln<<<256 * 180, 256, 0, stream>>>(G2, b2, Xc, nw, nb, Xo);
    float* t = Xc; Xc = Xo; Xo = t;
  };

  // Only the li = EL-1 = 1 chain survives in the reference (x_all[...,0] is
  // overwritten each iteration; x_all[...,1] stays zero).
  const long long TW = 180LL * 256 * 256, TB = 180LL * 256;
  const long long CW = 256LL * 180 * 180, CB = 256LL * 180;
  temp_mix(tmd_w1 + TW, tmd_b1 + TB, tmd_w2 + TW, tmd_b2 + TB, tmd_nw + 256, tmd_nb + 256);
  chan_mix(cmd_w1 + CW, cmd_b1 + CB, cmd_w2 + CW, cmd_b2 + CB, cmd_nw + 256, cmd_nb + 256);
  temp_mix(tmu_w1 + TW, tmu_b1 + TB, tmu_w2 + TW, tmu_b2 + TB, tmu_nw + 256, tmu_nb + 256);
  chan_mix(cmu_w1 + CW, cmu_b1 + CB, cmu_w2 + CW, cmu_b2 + CB, cmu_nw + 256, cmu_nb + 256);
  temp_mix(tmu_w1, tmu_b1, tmu_w2, tmu_b2, tmu_nw, tmu_nb);
  chan_mix(cmu_w1, cmu_b1, cmu_w2, cmu_b2, cmu_nw, cmu_nb);

  // ----- combine, s-correction, head -----
  k_ln_plain<<<256 * 90, 256, 0, stream>>>(xold, xoldln, 90, 90, lnw, lnb);
  k_comb<<<(unsigned)((11796480LL + 255) / 256), 256, 0, stream>>>(
      Xc, combw, combb, 11796480LL);
  k_ln_plain<<<256 * 90, 256, 0, stream>>>(Xc, xcln, 180, 90, lnw, lnb);
  k_scorr<<<65536, 128, 0, stream>>>(xoldln, xcln, scalep);
  k_scale_transpose<<<(unsigned)((11796480LL + 255) / 256), 256, 0, stream>>>(
      Xc, scalep, G1 /*xct: (bn, d*180+p)*/);

  // head: (256 x 46080) @ (46080 x 720->736 padded), split-K=10 for occupancy
  packA(G1, 1, 0, 46080, 1, 256, 46080, 16, 1440);
  packB(headw, 1, 0, 720, 1, 46080, 720, 1440, 46);
  gemm(G2, 1, 16, 46, 1440, 10, 144);
  k_reduce_splitk<<<(unsigned)((184320LL + 255) / 256), 256, 0, stream>>>(
      G2, headC, 10, 256, 736, 720);
  k_head_final<<<(unsigned)((184320LL + 255) / 256), 256, 0, stream>>>(
      headC, headb, revw, revb, meanp, stdp, (float*)d_out);
}
